// GRANMixtureBernoulli_79972291051527
// MI455X (gfx1250) — matrix operand order repaired and checked by reading, not compiled, not run on hardware
//
#include <hip/hip_runtime.h>
#include <hip/hip_bf16.h>

typedef __attribute__((ext_vector_type(16))) __bf16 v16bf;
typedef __attribute__((ext_vector_type(8)))  __bf16 v8bf;
typedef __attribute__((ext_vector_type(4)))  __bf16 v4bf;
typedef __attribute__((ext_vector_type(8)))  float  v8f;

#define N_GNN   32768
#define N_FEAT  32769
#define E_CNT   200000
#define E_OUT   500000
#define NUM_SUB 1024
#define K_MIX   20
#define DSZ     192
#define HID     128

__device__ __forceinline__ float bf2f(__bf16 v) { return (float)v; }
__device__ __forceinline__ __bf16 f2bf(float v) { return (__bf16)v; }

// ---- WMMA fragment helpers (CDNA5 16x16x32 bf16 layouts, wave32) ----
// A: 16(M) x 32(K). lane L<16: M=L, elems = K[koff..koff+7] ++ K[16+koff..16+koff+7], koff = (L>=16)?8:0
// Two contiguous 8x bf16 runs -> two 16-byte LDS loads.
__device__ __forceinline__ v16bf frag_a(const __bf16* tile, int ldk, int lane) {
  int row  = lane & 15;
  int koff = (lane & 16) ? 8 : 0;
  const __bf16* r = tile + (size_t)row * ldk + koff;
  union { v16bf v; v8bf h[2]; } u;
  u.h[0] = *(const v8bf*)(r);
  u.h[1] = *(const v8bf*)(r + 16);
  return u.v;
}
// B from TRANSPOSED weights Wt[N][K]: lane L<16: N=L, K=k0+[0..7]++[16..23]; lane>=16: N=L-16, K=k0+[8..15]++[24..31].
// Two runs of 8 contiguous K values -> two 16-byte global loads.
__device__ __forceinline__ v16bf frag_bT(const __bf16* Wt, int ldk, int k0, int n0, int lane) {
  int n  = lane & 15;
  int kb = (lane & 16) ? 8 : 0;
  const __bf16* r = Wt + (size_t)(n0 + n) * ldk + k0 + kb;
  union { v16bf v; v8bf h[2]; } u;
  u.h[0] = *(const v8bf*)(r);
  u.h[1] = *(const v8bf*)(r + 16);
  return u.v;
}
__device__ __forceinline__ v8f wmma_bf16(v16bf a, v16bf b, v8f c) {
  return __builtin_amdgcn_wmma_f32_16x16x32_bf16(false, a, false, b, (short)0, c, false, false);
}
// float4 -> 4 packed bf16
__device__ __forceinline__ v4bf pk4(float x, float y, float z, float w) {
  v4bf o; o[0] = f2bf(x); o[1] = f2bf(y); o[2] = f2bf(z); o[3] = f2bf(w);
  return o;
}

// ------------------- utility kernels -------------------
__global__ void zero_f32(float* p, size_t n) {
  size_t i = (size_t)blockIdx.x * blockDim.x + threadIdx.x;
  for (; i < n; i += (size_t)gridDim.x * blockDim.x) p[i] = 0.0f;
}
// transpose-convert: src f32 [K][N] -> dst bf16 [N][K]
__global__ void cvt_bf16_T(const float* s, __bf16* d, int K, int N) {
  int i = blockIdx.x * blockDim.x + threadIdx.x;
  if (i >= K * N) return;
  int k = i / N, n = i % N;
  d[(size_t)n * K + k] = f2bf(s[i]);
}
// transpose-convert with N padded (zero rows beyond N): src [K][N] -> dst [Npad][K]
__global__ void cvt_bf16_T_pad(const float* s, __bf16* d, int K, int N, int Npad) {
  int i = blockIdx.x * blockDim.x + threadIdx.x;
  if (i >= K * Npad) return;
  int k = i / Npad, n = i % Npad;
  d[(size_t)n * K + k] = (n < N) ? f2bf(s[(size_t)k * N + n]) : (__bf16)0.0f;
}

// ------------------- node_feat = A @ W_dec + b -------------------
// M=32768, K=1024, N=128. Block: 16 rows, 8 waves (one 16-col tile each), 16x64 A panel per sync.
__global__ __launch_bounds__(256) void nodefeat_gemm(const float* __restrict__ A,
                                                     const __bf16* __restrict__ WdecT,  // [128][1024]
                                                     const float* __restrict__ bdec,
                                                     float* __restrict__ nf) {
  __shared__ __align__(16) __bf16 At[16 * 64];
  int m0 = blockIdx.x * 16;
  int t = threadIdx.x, lane = t & 31, wave = t >> 5;
  int n0 = wave * 16;
  v8f acc = {};
  for (int k0 = 0; k0 < 1024; k0 += 64) {
    {  // 16x64 fp32 panel = 256 float4, one per thread
      int r = t >> 4, c = (t & 15) << 2;
      const float4 va = *(const float4*)&A[(size_t)(m0 + r) * 1024 + k0 + c];
      *(v4bf*)&At[r * 64 + c] = pk4(va.x, va.y, va.z, va.w);
    }
    if (k0 + 64 < 1024)
      __builtin_prefetch(&A[(size_t)(m0 + (t >> 4)) * 1024 + k0 + 64], 0, 0);
    __syncthreads();
    acc = wmma_bf16(frag_a(At, 64, lane),      frag_bT(WdecT, 1024, k0, n0, lane),      acc);
    acc = wmma_bf16(frag_a(At + 32, 64, lane), frag_bT(WdecT, 1024, k0 + 32, n0, lane), acc);
    __syncthreads();
  }
  int n = n0 + (lane & 15);
  int mb = (lane & 16) ? 8 : 0;
#pragma unroll
  for (int i = 0; i < 8; ++i)
    nf[(size_t)(m0 + mb + i + 1) * 128 + n] = acc[i] + bdec[n];  // row 0 reserved zero
}

// ------------------- state gather -------------------
__global__ void build_state(const float* __restrict__ nf, const int* __restrict__ nif,
                            const int* __restrict__ cls, const float* __restrict__ cemb,
                            float* __restrict__ state) {
  int i = blockIdx.x * blockDim.x + threadIdx.x;
  if (i >= N_GNN * DSZ) return;
  int m = i / DSZ, n = i % DSZ;
  state[i] = (n < HID) ? nf[(size_t)nif[m] * 128 + n] : cemb[cls[m] * 64 + (n - HID)];
}

// ------------------- edge propagation (msg & att MLPs + segment sum) -------------------
__global__ __launch_bounds__(256) void edge_prop(const float* __restrict__ state,
                                                 const int* __restrict__ edges,
                                                 const int* __restrict__ att_idx,
                                                 const __bf16* __restrict__ Wm1T, const float* __restrict__ bm1,  // [192][320]
                                                 const __bf16* __restrict__ Wm2T, const float* __restrict__ bm2,  // [192][192]
                                                 const __bf16* __restrict__ Wa1T, const float* __restrict__ ba1,  // [128][320]
                                                 const __bf16* __restrict__ Wa2T, const float* __restrict__ ba2,  // [192][128]
                                                 float* __restrict__ agg) {
  __shared__ __align__(16) __bf16 diff[16 * 192];
  __shared__ __align__(16) __bf16 h1[16 * 192];
  __shared__ __align__(16) __bf16 a1[16 * 128];
  __shared__ float  msg[16 * 192];
  __shared__ float  atl[16 * 192];
  __shared__ int sS[16], sD[16], sAs[16], sAd[16];
  int e0 = blockIdx.x * 16;
  int t = threadIdx.x, lane = t & 31, wave = t >> 5;
  if (t < 16) {
    int s = edges[(e0 + t) * 2 + 0], d = edges[(e0 + t) * 2 + 1];
    sS[t] = s; sD[t] = d; sAs[t] = att_idx[s]; sAd[t] = att_idx[d];
  }
  __syncthreads();
  for (int i4 = t; i4 < 16 * 48; i4 += 256) {          // vectorized diff build
    int m = i4 / 48, c = (i4 % 48) << 2;
    const float4 a = *(const float4*)&state[(size_t)sS[m] * DSZ + c];
    const float4 b = *(const float4*)&state[(size_t)sD[m] * DSZ + c];
    *(v4bf*)&diff[m * 192 + c] = pk4(a.x - b.x, a.y - b.y, a.z - b.z, a.w - b.w);
  }
  __syncthreads();
  // stage 1: jobs 0..11 msg hidden (192 cols), 12..19 att hidden (128 cols); K=192
  for (int job = wave; job < 20; job += 8) {
    v8f acc = {};
    if (job < 12) {
      int n0 = job * 16;
      for (int k0 = 0; k0 < 192; k0 += 32)
        acc = wmma_bf16(frag_a(diff + k0, 192, lane), frag_bT(Wm1T, 320, k0, n0, lane), acc);
      int n = n0 + (lane & 15), mb = (lane & 16) ? 8 : 0;
#pragma unroll
      for (int i = 0; i < 8; ++i) {
        int m = mb + i;
        float v = acc[i] + bm1[n] + bf2f(Wm1T[(size_t)n * 320 + 192 + sAs[m]]) +
                  bf2f(Wm1T[(size_t)n * 320 + 256 + sAd[m]]);      // one-hot edge features
        h1[m * 192 + n] = f2bf(fmaxf(v, 0.0f));
      }
    } else {
      int n0 = (job - 12) * 16;
      for (int k0 = 0; k0 < 192; k0 += 32)
        acc = wmma_bf16(frag_a(diff + k0, 192, lane), frag_bT(Wa1T, 320, k0, n0, lane), acc);
      int n = n0 + (lane & 15), mb = (lane & 16) ? 8 : 0;
#pragma unroll
      for (int i = 0; i < 8; ++i) {
        int m = mb + i;
        float v = acc[i] + ba1[n] + bf2f(Wa1T[(size_t)n * 320 + 192 + sAs[m]]) +
                  bf2f(Wa1T[(size_t)n * 320 + 256 + sAd[m]]);
        a1[m * 128 + n] = f2bf(fmaxf(v, 0.0f));
      }
    }
  }
  __syncthreads();
  // stage 2: jobs 0..11 msg out (K=192), 12..23 att out (K=128)
  for (int job = wave; job < 24; job += 8) {
    v8f acc = {};
    int n0 = (job % 12) * 16;
    if (job < 12) {
      for (int k0 = 0; k0 < 192; k0 += 32)
        acc = wmma_bf16(frag_a(h1 + k0, 192, lane), frag_bT(Wm2T, 192, k0, n0, lane), acc);
      int n = n0 + (lane & 15), mb = (lane & 16) ? 8 : 0;
#pragma unroll
      for (int i = 0; i < 8; ++i) msg[(mb + i) * 192 + n] = acc[i] + bm2[n];
    } else {
      for (int k0 = 0; k0 < 128; k0 += 32)
        acc = wmma_bf16(frag_a(a1 + k0, 128, lane), frag_bT(Wa2T, 128, k0, n0, lane), acc);
      int n = n0 + (lane & 15), mb = (lane & 16) ? 8 : 0;
#pragma unroll
      for (int i = 0; i < 8; ++i) atl[(mb + i) * 192 + n] = acc[i] + ba2[n];
    }
  }
  __syncthreads();
  for (int i = t; i < 16 * 192; i += 256) {
    int m = i / 192, n = i % 192;
    float p = msg[i] * (1.0f / (1.0f + __expf(-atl[i])));
    atomicAdd(&agg[(size_t)sD[m] * DSZ + n], p);       // segment_sum over dst
  }
}

// ------------------- GRU update -------------------
__global__ __launch_bounds__(256) void gru_update(const float* __restrict__ agg,
                                                  float* __restrict__ state,
                                                  const __bf16* __restrict__ WihT, const float* __restrict__ bih,  // [576][192]
                                                  const __bf16* __restrict__ WhhT, const float* __restrict__ bhh) {
  __shared__ __align__(16) __bf16 aT[16 * 192];
  __shared__ __align__(16) __bf16 sT[16 * 192];
  __shared__ float gi[16 * 576];
  __shared__ float gh[16 * 576];
  int m0 = blockIdx.x * 16;
  int t = threadIdx.x, lane = t & 31, wave = t >> 5;
  for (int i4 = t; i4 < 16 * 48; i4 += 256) {
    int m = i4 / 48, c = (i4 % 48) << 2;
    const float4 a = *(const float4*)&agg[(size_t)(m0 + m) * DSZ + c];
    const float4 s = *(const float4*)&state[(size_t)(m0 + m) * DSZ + c];
    *(v4bf*)&aT[m * 192 + c] = pk4(a.x, a.y, a.z, a.w);
    *(v4bf*)&sT[m * 192 + c] = pk4(s.x, s.y, s.z, s.w);
  }
  __syncthreads();
  for (int job = wave; job < 72; job += 8) {           // 36 gi tiles + 36 gh tiles
    v8f acc = {};
    int n0 = (job % 36) * 16;
    const __bf16* Am = (job < 36) ? aT : sT;
    const __bf16* W  = (job < 36) ? WihT : WhhT;
    for (int k0 = 0; k0 < 192; k0 += 32)
      acc = wmma_bf16(frag_a(Am + k0, 192, lane), frag_bT(W, 192, k0, n0, lane), acc);
    float* G = (job < 36) ? gi : gh;
    int n = n0 + (lane & 15), mb = (lane & 16) ? 8 : 0;
#pragma unroll
    for (int i = 0; i < 8; ++i) G[(mb + i) * 576 + n] = acc[i];
  }
  __syncthreads();
  for (int i = t; i < 16 * 192; i += 256) {
    int m = i / 192, n = i % 192;
    float ir = gi[m * 576 + n] + bih[n],                   hr = gh[m * 576 + n] + bhh[n];
    float iz = gi[m * 576 + 192 + n] + bih[192 + n],       hz = gh[m * 576 + 192 + n] + bhh[192 + n];
    float ii = gi[m * 576 + 384 + n] + bih[384 + n],       hn = gh[m * 576 + 384 + n] + bhh[384 + n];
    float r = 1.0f / (1.0f + __expf(-(ir + hr)));
    float z = 1.0f / (1.0f + __expf(-(iz + hz)));
    float nn = tanhf(ii + r * hn);
    size_t idx = (size_t)(m0 + m) * DSZ + n;
    state[idx] = (1.0f - z) * nn + z * state[idx];
  }
}

// ------------------- output MLPs + per-subgraph reductions -------------------
__global__ __launch_bounds__(256) void out_mlp(const float* __restrict__ state,
                                               const int* __restrict__ nig,
                                               const float* __restrict__ label,
                                               const int* __restrict__ subidx,
                                               const __bf16* __restrict__ Wt1T, const float* __restrict__ bt1,  // [128][192]
                                               const __bf16* __restrict__ Wt2T, const float* __restrict__ bt2,  // [128][128]
                                               const __bf16* __restrict__ Wt3T, const float* __restrict__ bt3,  // [32][128] padded
                                               const __bf16* __restrict__ Wa1T, const float* __restrict__ ba1,
                                               const __bf16* __restrict__ Wa2T, const float* __restrict__ ba2,
                                               const __bf16* __restrict__ Wa3T, const float* __restrict__ ba3,
                                               float* __restrict__ red_loss, float* __restrict__ red_alpha,
                                               float* __restrict__ cnt) {
  __shared__ __align__(16) __bf16 dout[16 * 192];
  __shared__ __align__(16) __bf16 h1[16 * 128], g1[16 * 128];
  __shared__ __align__(16) __bf16 h2[16 * 128], g2[16 * 128];
  __shared__ float lt[16 * 32], la[16 * 32];
  int p0 = blockIdx.x * 16;
  int t = threadIdx.x, lane = t & 31, wave = t >> 5;
  for (int i4 = t; i4 < 16 * 48; i4 += 256) {
    int m = i4 / 48, c = (i4 % 48) << 2;
    int i0 = nig[(p0 + m) * 2], i1 = nig[(p0 + m) * 2 + 1];
    const float4 a = *(const float4*)&state[(size_t)i0 * DSZ + c];
    const float4 b = *(const float4*)&state[(size_t)i1 * DSZ + c];
    *(v4bf*)&dout[m * 192 + c] = pk4(a.x - b.x, a.y - b.y, a.z - b.z, a.w - b.w);
  }
  __syncthreads();
  for (int job = wave; job < 16; job += 8) {           // stage 1, K=192
    v8f acc = {};
    int n0 = (job & 7) * 16;
    const __bf16* W = (job < 8) ? Wt1T : Wa1T;
    const float*  b = (job < 8) ? bt1 : ba1;
    for (int k0 = 0; k0 < 192; k0 += 32)
      acc = wmma_bf16(frag_a(dout + k0, 192, lane), frag_bT(W, 192, k0, n0, lane), acc);
    __bf16* O = (job < 8) ? h1 : g1;
    int n = n0 + (lane & 15), mb = (lane & 16) ? 8 : 0;
#pragma unroll
    for (int i = 0; i < 8; ++i) O[(mb + i) * 128 + n] = f2bf(fmaxf(acc[i] + b[n], 0.0f));
  }
  __syncthreads();
  for (int job = wave; job < 16; job += 8) {           // stage 2, K=128
    v8f acc = {};
    int n0 = (job & 7) * 16;
    const __bf16* Am = (job < 8) ? h1 : g1;
    const __bf16* W  = (job < 8) ? Wt2T : Wa2T;
    const float*  b  = (job < 8) ? bt2 : ba2;
    for (int k0 = 0; k0 < 128; k0 += 32)
      acc = wmma_bf16(frag_a(Am + k0, 128, lane), frag_bT(W, 128, k0, n0, lane), acc);
    __bf16* O = (job < 8) ? h2 : g2;
    int n = n0 + (lane & 15), mb = (lane & 16) ? 8 : 0;
#pragma unroll
    for (int i = 0; i < 8; ++i) O[(mb + i) * 128 + n] = f2bf(fmaxf(acc[i] + b[n], 0.0f));
  }
  __syncthreads();
  for (int job = wave; job < 4; job += 8) {            // stage 3 (20-wide padded to 32), K=128
    v8f acc = {};
    int n0 = (job & 1) * 16;
    const __bf16* Am = (job < 2) ? h2 : g2;
    const __bf16* W  = (job < 2) ? Wt3T : Wa3T;
    for (int k0 = 0; k0 < 128; k0 += 32)
      acc = wmma_bf16(frag_a(Am + k0, 128, lane), frag_bT(W, 128, k0, n0, lane), acc);
    float* O = (job < 2) ? lt : la;
    int n = n0 + (lane & 15), mb = (lane & 16) ? 8 : 0;
#pragma unroll
    for (int i = 0; i < 8; ++i) O[(mb + i) * 32 + n] = acc[i];
  }
  __syncthreads();
  if (t < 16) {
    int m = t, pid = p0 + m;
    int sg = subidx[pid];
    float lab = label[pid];
    float av[K_MIX], mx = -1e30f;
#pragma unroll
    for (int k = 0; k < K_MIX; ++k) { av[k] = la[m * 32 + k] + ba3[k]; mx = fmaxf(mx, av[k]); }
    float se = 0.0f;
#pragma unroll
    for (int k = 0; k < K_MIX; ++k) se += __expf(av[k] - mx);
    float lse = mx + __logf(se);
#pragma unroll
    for (int k = 0; k < K_MIX; ++k) {
      float th = lt[m * 32 + k] + bt3[k];
      float sp = fmaxf(th, 0.0f) + log1pf(__expf(-fabsf(th)));   // stable softplus
      atomicAdd(&red_loss[sg * K_MIX + k], sp - th * lab);
      atomicAdd(&red_alpha[sg * K_MIX + k], av[k] - lse);        // log_softmax
    }
    atomicAdd(&cnt[sg], 1.0f);
  }
}

// ------------------- final logsumexp + sum -------------------
__global__ __launch_bounds__(1024) void final_reduce(const float* __restrict__ red_loss,
                                                     const float* __restrict__ red_alpha,
                                                     const float* __restrict__ cnt,
                                                     float* __restrict__ out) {
  __shared__ float sb[1024];
  int s = threadIdx.x;
  float c = fmaxf(cnt[s], 1.0f);
  float mx = -1e30f, v[K_MIX];
#pragma unroll
  for (int k = 0; k < K_MIX; ++k) {
    v[k] = -red_loss[s * K_MIX + k] + red_alpha[s * K_MIX + k] / c;
    mx = fmaxf(mx, v[k]);
  }
  float se = 0.0f;
#pragma unroll
  for (int k = 0; k < K_MIX; ++k) se += __expf(v[k] - mx);
  sb[s] = mx + __logf(se);
  __syncthreads();
  for (int off = 512; off > 0; off >>= 1) {
    if (s < off) sb[s] += sb[s + off];
    __syncthreads();
  }
  if (s == 0) out[0] = -sb[0] / (float)E_OUT;
}

// ------------------- host orchestration -------------------
extern "C" void kernel_launch(void* const* d_in, const int* in_sizes, int n_in,
                              void* d_out, int out_size, void* d_ws, size_t ws_size,
                              hipStream_t stream) {
  const float* A      = (const float*)d_in[0];
  const int* edges    = (const int*)d_in[1];
  const int* att_idx  = (const int*)d_in[2];
  const int* nif      = (const int*)d_in[3];
  const int* cls      = (const int*)d_in[4];
  const int* nig      = (const int*)d_in[5];
  const float* label  = (const float*)d_in[6];
  const int* subidx   = (const int*)d_in[7];
  const float* Wdec = (const float*)d_in[8],  *bdec = (const float*)d_in[9];
  const float* cemb = (const float*)d_in[10];
  const float* Wm1  = (const float*)d_in[11], *bm1 = (const float*)d_in[12];
  const float* Wm2  = (const float*)d_in[13], *bm2 = (const float*)d_in[14];
  const float* Wa1  = (const float*)d_in[15], *ba1 = (const float*)d_in[16];
  const float* Wa2  = (const float*)d_in[17], *ba2 = (const float*)d_in[18];
  const float* Wih  = (const float*)d_in[19], *bih = (const float*)d_in[20];
  const float* Whh  = (const float*)d_in[21], *bhh = (const float*)d_in[22];
  const float* Wt1  = (const float*)d_in[23], *bt1 = (const float*)d_in[24];
  const float* Wt2  = (const float*)d_in[25], *bt2 = (const float*)d_in[26];
  const float* Wt3  = (const float*)d_in[27], *bt3 = (const float*)d_in[28];
  const float* Wo1  = (const float*)d_in[29], *bo1 = (const float*)d_in[30];
  const float* Wo2  = (const float*)d_in[31], *bo2 = (const float*)d_in[32];
  const float* Wo3  = (const float*)d_in[33], *bo3 = (const float*)d_in[34];

  char* ws = (char*)d_ws;
  size_t off = 0;
  auto alloc = [&](size_t bytes) { size_t r = off; off = (off + bytes + 255) & ~(size_t)255; return r; };
  float* state    = (float*)(ws + alloc((size_t)N_GNN * DSZ * 4));
  float* agg      = (float*)(ws + alloc((size_t)N_GNN * DSZ * 4));
  float* nf       = (float*)(ws + alloc((size_t)N_FEAT * 128 * 4));
  float* red_loss = (float*)(ws + alloc((size_t)NUM_SUB * K_MIX * 4));
  float* red_alp  = (float*)(ws + alloc((size_t)NUM_SUB * K_MIX * 4));
  float* cnt      = (float*)(ws + alloc((size_t)NUM_SUB * 4));
  __bf16* bWdecT = (__bf16*)(ws + alloc((size_t)128 * 1024 * 2));
  __bf16* bWm1T  = (__bf16*)(ws + alloc((size_t)192 * 320 * 2));
  __bf16* bWm2T  = (__bf16*)(ws + alloc((size_t)192 * 192 * 2));
  __bf16* bWa1T  = (__bf16*)(ws + alloc((size_t)128 * 320 * 2));
  __bf16* bWa2T  = (__bf16*)(ws + alloc((size_t)192 * 128 * 2));
  __bf16* bWihT  = (__bf16*)(ws + alloc((size_t)576 * 192 * 2));
  __bf16* bWhhT  = (__bf16*)(ws + alloc((size_t)576 * 192 * 2));
  __bf16* bWt1T  = (__bf16*)(ws + alloc((size_t)128 * 192 * 2));
  __bf16* bWt2T  = (__bf16*)(ws + alloc((size_t)128 * 128 * 2));
  __bf16* bWt3T  = (__bf16*)(ws + alloc((size_t)32 * 128 * 2));   // padded 20->32 rows
  __bf16* bWo1T  = (__bf16*)(ws + alloc((size_t)128 * 192 * 2));
  __bf16* bWo2T  = (__bf16*)(ws + alloc((size_t)128 * 128 * 2));
  __bf16* bWo3T  = (__bf16*)(ws + alloc((size_t)32 * 128 * 2));

  auto CVT = [&](const float* s, __bf16* d, int K, int N) {
    cvt_bf16_T<<<(K * N + 255) / 256, 256, 0, stream>>>(s, d, K, N);
  };
  CVT(Wdec, bWdecT, 1024, 128);
  CVT(Wm1,  bWm1T,  320, 192);
  CVT(Wm2,  bWm2T,  192, 192);
  CVT(Wa1,  bWa1T,  320, 128);
  CVT(Wa2,  bWa2T,  128, 192);
  CVT(Wih,  bWihT,  192, 576);
  CVT(Whh,  bWhhT,  192, 576);
  CVT(Wt1,  bWt1T,  192, 128);
  CVT(Wt2,  bWt2T,  128, 128);
  CVT(Wo1,  bWo1T,  192, 128);
  CVT(Wo2,  bWo2T,  128, 128);
  cvt_bf16_T_pad<<<(128 * 32 + 255) / 256, 256, 0, stream>>>(Wt3, bWt3T, 128, 20, 32);
  cvt_bf16_T_pad<<<(128 * 32 + 255) / 256, 256, 0, stream>>>(Wo3, bWo3T, 128, 20, 32);

  zero_f32<<<1, 128, 0, stream>>>(nf, 128);                       // zero row 0
  nodefeat_gemm<<<N_GNN / 16, 256, 0, stream>>>(A, bWdecT, bdec, nf);
  build_state<<<(N_GNN * DSZ + 255) / 256, 256, 0, stream>>>(nf, nif, cls, cemb, state);

  for (int p = 0; p < 2; ++p) {
    zero_f32<<<2048, 256, 0, stream>>>(agg, (size_t)N_GNN * DSZ);
    edge_prop<<<E_CNT / 16, 256, 0, stream>>>(state, edges, att_idx,
                                              bWm1T, bm1, bWm2T, bm2, bWa1T, ba1, bWa2T, ba2, agg);
    gru_update<<<N_GNN / 16, 256, 0, stream>>>(agg, state, bWihT, bih, bWhhT, bhh);
  }

  zero_f32<<<80, 256, 0, stream>>>(red_loss, (size_t)NUM_SUB * K_MIX);
  zero_f32<<<80, 256, 0, stream>>>(red_alp,  (size_t)NUM_SUB * K_MIX);
  zero_f32<<<4, 256, 0, stream>>>(cnt, (size_t)NUM_SUB);

  out_mlp<<<E_OUT / 16, 256, 0, stream>>>(state, nig, label, subidx,
                                          bWt1T, bt1, bWt2T, bt2, bWt3T, bt3,
                                          bWo1T, bo1, bWo2T, bo2, bWo3T, bo3,
                                          red_loss, red_alp, cnt);
  final_reduce<<<1, 1024, 0, stream>>>(red_loss, red_alp, cnt, (float*)d_out);
}
